// GNNAggregationWithAttention_6055903887912
// MI455X (gfx1250) — matrix operand order repaired and checked by reading, not compiled, not run on hardware
//
#include <hip/hip_runtime.h>

#define NN     10000                 // nodes
#define DD     256                   // embedding dim
#define BB     8192                  // batch
#define NPAD   10112                 // N padded to multiple of 128 (79*128)
#define NSTAGE (NPAD / 128)          // 79 K-stages of 128

typedef __attribute__((ext_vector_type(16))) __bf16 v16bf;
typedef __attribute__((ext_vector_type(8)))  float  v8f;
typedef __attribute__((ext_vector_type(8)))  unsigned short us8;

union Frag16 { us8 h[2]; v16bf v; };

static __device__ __forceinline__ unsigned short f32_to_bf16(float f) {
    unsigned int u = __float_as_uint(f);
    u += 0x7FFFu + ((u >> 16) & 1u);    // round-to-nearest-even
    return (unsigned short)(u >> 16);
}

// ---------------- Prep 1: E (N x D, f32 row-major) -> Et (D x NPAD, bf16), zero-padded
__global__ __launch_bounds__(256) void gnn_prep_et(const float* __restrict__ E,
                                                   unsigned short* __restrict__ Et) {
    __shared__ float tile[32][33];
    const int tx = threadIdx.x;           // 0..31
    const int ty = threadIdx.y;           // 0..7
    const int bn = blockIdx.x;            // 0..315 (n tiles)
    const int bd = blockIdx.y;            // 0..7   (d tiles)
    #pragma unroll
    for (int r = 0; r < 4; ++r) {
        const int n = bn * 32 + ty + r * 8;
        const int d = bd * 32 + tx;
        tile[ty + r * 8][tx] = (n < NN) ? E[(size_t)n * DD + d] : 0.0f;  // coalesced over d
    }
    __syncthreads();
    #pragma unroll
    for (int r = 0; r < 4; ++r) {
        const int d = bd * 32 + ty + r * 8;
        const int n = bn * 32 + tx;
        Et[(size_t)d * NPAD + n] = f32_to_bf16(tile[tx][ty + r * 8]);    // coalesced over n
    }
}

// ---------------- Prep 2: snb[n] = E[n].w_n + bias ; sc[b] = E[idx[b]].w_c
__global__ __launch_bounds__(256) void gnn_scores(const float* __restrict__ E,
                                                  const float* __restrict__ fcw,
                                                  const float* __restrict__ fcb,
                                                  const int* __restrict__ idx,
                                                  float* __restrict__ snb,
                                                  float* __restrict__ sc) {
    __shared__ float red[256];
    const int blk = blockIdx.x;
    const int t = threadIdx.x;
    float v;
    if (blk < NN) {
        v = E[(size_t)blk * DD + t] * fcw[DD + t];        // w_n
    } else {
        const int b = blk - NN;
        v = E[(size_t)idx[b] * DD + t] * fcw[t];          // w_c
    }
    red[t] = v;
    __syncthreads();
    #pragma unroll
    for (int s = 128; s > 0; s >>= 1) {
        if (t < s) red[t] += red[t + s];
        __syncthreads();
    }
    if (t == 0) {
        if (blk < NN) snb[blk] = red[0] + fcb[0];         // fold bias into s_n
        else          sc[blk - NN] = red[0];
    }
}

// ---------------- Main: out[b,:] = (mask*score) @ E + center, via bf16 WMMA, f32 accum
__global__ __launch_bounds__(256) void gnn_main(const int* __restrict__ idx,
                                                const float* __restrict__ A,
                                                const float* __restrict__ E,
                                                const unsigned short* __restrict__ Et,
                                                const float* __restrict__ snb,
                                                const float* __restrict__ sc,
                                                float* __restrict__ out) {
    __shared__ __align__(16) unsigned short P[16 * 128];   // 16(M) x 128(K) bf16 scores, 4 KB

    const int t    = threadIdx.x;
    const int tile = blockIdx.x;          // b-tile: rows tile*16 .. tile*16+15
    const int lane = t & 31;
    const int wave = t >> 5;              // 8 waves
    const int hl   = (lane >> 4) & 1;     // lane half
    const int ln   = lane & 15;

    // P-build role: thread t computes P[pm][pkb .. pkb+7]
    const int  pm  = t >> 4;
    const int  pkb = (t & 15) << 3;       // 8 consecutive k per thread
    const int  pb  = tile * 16 + pm;
    const long long arow = (long long)idx[pb] * (long long)NN;   // 16B-aligned row (40000 % 16 == 0)
    const float scv = sc[pb];

    // WMMA role: this wave owns d columns [wave*32, wave*32+32)
    const int d0 = wave * 32;
    const unsigned short* etp0 = Et + (size_t)(d0 + ln) * NPAD + hl * 16;        // d-tile 0
    const unsigned short* etp1 = Et + (size_t)(d0 + 16 + ln) * NPAD + hl * 16;   // d-tile 1

    v8f acc0 = {};
    v8f acc1 = {};

    for (int st = 0; st < NSTAGE; ++st) {
        const int n0 = st * 128;

        __syncthreads();   // previous P fully consumed
        float pv[8];
        if (n0 + 127 < NN) {                       // uniform fast path (all but last stage)
            const float4 m0 = *(const float4*)(A + arow + n0 + pkb);
            const float4 m1 = *(const float4*)(A + arow + n0 + pkb + 4);
            const float4 s0 = *(const float4*)(snb + n0 + pkb);
            const float4 s1 = *(const float4*)(snb + n0 + pkb + 4);
            const float ms[8] = {m0.x, m0.y, m0.z, m0.w, m1.x, m1.y, m1.z, m1.w};
            const float ss[8] = {s0.x, s0.y, s0.z, s0.w, s1.x, s1.y, s1.z, s1.w};
            #pragma unroll
            for (int j = 0; j < 8; ++j) {
                float s = scv + ss[j];                       // bias already folded in
                s = (s > 0.0f) ? s : 0.2f * s;               // leaky_relu(0.2)
                pv[j] = (ms[j] > 0.0f) ? s : 0.0f;           // mask
            }
            if (n0 + pkb + 1024 < NN)
                __builtin_prefetch(A + arow + n0 + pkb + 1024, 0, 1);  // global_prefetch_b8
        } else {                                   // uniform tail path (last stage only)
            #pragma unroll
            for (int j = 0; j < 8; ++j) {
                const int n = n0 + pkb + j;
                float p = 0.0f;
                if (n < NN) {
                    float s = scv + snb[n];
                    s = (s > 0.0f) ? s : 0.2f * s;
                    p = (A[arow + n] > 0.0f) ? s : 0.0f;
                }
                pv[j] = p;
            }
        }
        us8 pk8;
        #pragma unroll
        for (int j = 0; j < 8; ++j) pk8[j] = f32_to_bf16(pv[j]);
        *(us8*)&P[pm * 128 + pkb] = pk8;           // one ds_store_b128
        __syncthreads();   // P ready

        // 4 K-chunks of 32, 2 WMMA each
        #pragma unroll
        for (int c = 0; c < 4; ++c) {
            const int ko = c * 32;
            // A-fragment (16x32 bf16): lane ln holds M=ln; K = hl*16 + 0..15
            Frag16 af;
            af.h[0] = *(const us8*)&P[ln * 128 + ko + hl * 16];
            af.h[1] = *(const us8*)&P[ln * 128 + ko + hl * 16 + 8];
            // B-fragments (32x16 bf16): lane ln holds N=ln; K = hl*16 + 0..15, contiguous in Et
            Frag16 b0, b1;
            b0.h[0] = *(const us8*)(etp0 + n0 + ko);
            b0.h[1] = *(const us8*)(etp0 + n0 + ko + 8);
            b1.h[0] = *(const us8*)(etp1 + n0 + ko);
            b1.h[1] = *(const us8*)(etp1 + n0 + ko + 8);

            acc0 = __builtin_amdgcn_wmma_f32_16x16x32_bf16(false, af.v, false, b0.v,
                                                           (short)0, acc0, false, false);
            acc1 = __builtin_amdgcn_wmma_f32_16x16x32_bf16(false, af.v, false, b1.v,
                                                           (short)0, acc1, false, false);
        }
    }

    // Epilogue: C/D layout — VGPR r holds (M = r + 8*hl, N = ln); add center, store
    #pragma unroll
    for (int r = 0; r < 8; ++r) {
        const int b = tile * 16 + r + hl * 8;
        const long long cb = (long long)idx[b] * (long long)DD;
        const int dA = d0 + ln;
        const int dB = d0 + 16 + ln;
        out[(size_t)b * DD + dA] = acc0[r] + E[cb + dA];
        out[(size_t)b * DD + dB] = acc1[r] + E[cb + dB];
    }
}

extern "C" void kernel_launch(void* const* d_in, const int* in_sizes, int n_in,
                              void* d_out, int out_size, void* d_ws, size_t ws_size,
                              hipStream_t stream) {
    const int*   idx = (const int*)  d_in[0];   // node_indexes (B,)
    const float* A   = (const float*)d_in[1];   // A (N,N)
    const float* E   = (const float*)d_in[2];   // embedding_states (N,D)
    const float* fcw = (const float*)d_in[3];   // fc_w (1,2D)
    const float* fcb = (const float*)d_in[4];   // fc_b (1,)
    float*       out = (float*)d_out;           // (B,D)

    // workspace layout: Et (D x NPAD bf16) | snb (NPAD f32) | sc (B f32)  ~= 5.3 MB
    unsigned short* Et = (unsigned short*)d_ws;
    float* snb = (float*)((char*)d_ws + (size_t)DD * NPAD * sizeof(unsigned short));
    float* sc  = snb + NPAD;

    gnn_prep_et<<<dim3(NPAD / 32, DD / 32), dim3(32, 8), 0, stream>>>(E, Et);
    gnn_scores<<<NN + BB, 256, 0, stream>>>(E, fcw, fcb, idx, snb, sc);
    gnn_main<<<BB / 16, 256, 0, stream>>>(idx, A, E, Et, snb, sc, out);
}
